// MPNN_5368709120093
// MI455X (gfx1250) — compile-verified, matrix operand order checked by reading, not run on hardware
//
#include <hip/hip_runtime.h>

#define N_NODES 15000
#define N_EDGES 60000
#define NBATCH  512
#define DIM     64
#define NOUT    12
#define NI      15
#define EI_DIM  5
#define EH      128
#define STEPS   6

typedef __attribute__((ext_vector_type(16))) __bf16 v16bf;
typedef __attribute__((ext_vector_type(8)))  float  v8f;

// ---------- helpers ----------
__device__ __forceinline__ unsigned short f32_to_bf16(float f) {
    unsigned u = __float_as_uint(f);
    unsigned r = u + 0x7FFFu + ((u >> 16) & 1u);   // round-to-nearest-even
    return (unsigned short)(r >> 16);
}
__device__ __forceinline__ float bf16lo(unsigned w) { return __uint_as_float(w << 16); }
__device__ __forceinline__ float bf16hi(unsigned w) { return __uint_as_float(w & 0xFFFF0000u); }
__device__ __forceinline__ float sigm(float x) { return 1.0f / (1.0f + __expf(-x)); }

// order-preserving float<->uint for atomicMax-based segment max
__device__ __forceinline__ unsigned f32_ord(float f) {
    unsigned u = __float_as_uint(f);
    return (u & 0x80000000u) ? ~u : (u | 0x80000000u);
}
__device__ __forceinline__ float ord_f32(unsigned u) {
    return __uint_as_float((u & 0x80000000u) ? (u ^ 0x80000000u) : ~u);
}

// WMMA A/B packed index (16x32 bf16 tile): lane holds m (or n) in 0..15, half picks K sub-block.
__device__ __forceinline__ int pack_idx(int tile, int kc, int m, int kk) {
    int half = (kk >> 3) & 1;
    int idx  = (kk & 7) + ((kk & 16) ? 8 : 0);
    int lane = m + (half << 4);
    return ((tile * 4 + kc) * 32 + lane) * 16 + idx;
}

// ---------- fills ----------
__global__ void k_fill_f32(float* p, float v, int n) {
    int t = blockIdx.x * blockDim.x + threadIdx.x;
    if (t < n) p[t] = v;
}
__global__ void k_fill_u32(unsigned* p, unsigned v, int n) {
    int t = blockIdx.x * blockDim.x + threadIdx.x;
    if (t < n) p[t] = v;
}

// ---------- one-time weight packing ----------
// B pack: Bmat[k][n] = en_w2[n][k]  (en_w2 is [4096][128])
__global__ void k_pack_w2(const float* __restrict__ en_w2, unsigned short* __restrict__ w2pack) {
    int t = blockIdx.x * blockDim.x + threadIdx.x;   // 4096*128
    int n = t >> 7, k = t & 127;
    float v = en_w2[n * 128 + k];
    w2pack[pack_idx(n >> 4, k >> 5, n & 15, k & 31)] = f32_to_bf16(v);
}

// generic transpose: w[J][K] -> wT[K][J]
__global__ void k_transpose(const float* __restrict__ w, float* __restrict__ wT, int J, int K) {
    int t = blockIdx.x * blockDim.x + threadIdx.x;
    if (t >= J * K) return;
    int j = t / K, k = t % K;
    wT[k * J + j] = w[t];
}

// ---------- edge network: hid = relu(edge_attr@W1^T + b1), packed as WMMA A ----------
__global__ void k_hid_pack(const float* __restrict__ ea, const float* __restrict__ w1,
                           const float* __restrict__ b1, unsigned short* __restrict__ hidpack) {
    int t = blockIdx.x * blockDim.x + threadIdx.x;   // E*128
    int e = t >> 7, k = t & 127;
    float acc = b1[k];
#pragma unroll
    for (int c = 0; c < EI_DIM; ++c) acc += ea[e * EI_DIM + c] * w1[k * EI_DIM + c];
    acc = fmaxf(acc, 0.0f);
    hidpack[pack_idx(e >> 4, k >> 5, e & 15, k & 31)] = f32_to_bf16(acc);
}

// ---------- WMMA GEMM: ew[E][4096] (bf16) = hid[E][128] @ W2^T[128][4096] + b2 ----------
__global__ void k_gemm_ew(const unsigned short* __restrict__ hidpack,
                          const unsigned short* __restrict__ w2pack,
                          const float* __restrict__ en_b2,
                          unsigned short* __restrict__ ew) {
    __shared__ float cs[8][16][16];
    const int wave = threadIdx.x >> 5, lane = threadIdx.x & 31;
    const int mtile = blockIdx.x;                  // 0..3749
    const int ntile = blockIdx.y * 8 + wave;       // 0..255
    const v16bf* ap = (const v16bf*)hidpack;
    const v16bf* bp = (const v16bf*)w2pack;
    v8f c = {};
#pragma unroll
    for (int kc = 0; kc < 4; ++kc) {
        v16bf a = ap[(mtile * 4 + kc) * 32 + lane];
        v16bf b = bp[(ntile * 4 + kc) * 32 + lane];
        c = __builtin_amdgcn_wmma_f32_16x16x32_bf16(false, a, false, b, (short)0, c, false, false);
    }
    const float bias = en_b2[ntile * 16 + (lane & 15)];
    const int half = lane >> 4;
#pragma unroll
    for (int r = 0; r < 8; ++r) cs[wave][r + 8 * half][lane & 15] = c[r] + bias;
    __syncthreads();
    // coalesced bf16 row store: 2 lanes cover one 16-col row (32 bytes)
    const int row = lane >> 1, cb = (lane & 1) * 8;
    unsigned short o[8];
#pragma unroll
    for (int j = 0; j < 8; ++j) o[j] = f32_to_bf16(cs[wave][row][cb + j]);
    size_t off = (size_t)(mtile * 16 + row) * 4096 + ntile * 16 + cb;
    *(uint4*)(ew + off) = *(const uint4*)o;
}

// ---------- lin0: out = relu(x @ lin0_w^T + b) ----------
__global__ void k_lin0(const float* __restrict__ x, const float* __restrict__ w,
                       const float* __restrict__ b, float* __restrict__ h) {
    int t = blockIdx.x * blockDim.x + threadIdx.x;
    if (t >= N_NODES * DIM) return;
    int n = t >> 6, d = t & 63;
    float acc = b[d];
#pragma unroll
    for (int c = 0; c < NI; ++c) acc += x[n * NI + c] * w[d * NI + c];
    h[t] = fmaxf(acc, 0.0f);
}

// ---------- degree ----------
__global__ void k_deg(const int* __restrict__ ei, float* __restrict__ deg) {
    int t = blockIdx.x * blockDim.x + threadIdx.x;
    if (t < N_EDGES) atomicAdd(&deg[ei[N_EDGES + t]], 1.0f);
}

// ---------- MP: per-edge matvec msg = x_src @ ew[e] ; scatter-add into agg[tgt] ----------
// One wave32 per edge. Lane L: row-group rgrp = L>>3 (rows it*4+rgrp), 8-col slab cb = (L&7)*8.
// Each iteration: one global_load_b128 per lane (wave reads 4 full rows = 512B, fully coalesced).
// Cross-row reduction via two shfl_xor steps; lanes 0..7 scatter 64 f32 atomics per edge.
__global__ void k_edge_matvec(const unsigned short* __restrict__ ew, const float* __restrict__ out,
                              const int* __restrict__ ei, float* __restrict__ agg) {
    __shared__ float xs[8][64];
    const int wave = threadIdx.x >> 5, lane = threadIdx.x & 31;
    const int e = blockIdx.x * 8 + wave;          // E = 60000 = 7500*8, exact
    const int src = ei[e], tgt = ei[N_EDGES + e];
    xs[wave][lane]      = out[(size_t)src * 64 + lane];
    xs[wave][lane + 32] = out[(size_t)src * 64 + 32 + lane];
    __syncthreads();

    const int rgrp = lane >> 3;          // 0..3
    const int cb   = (lane & 7) * 8;     // 0,8,..,56
    const unsigned short* wp = ew + (size_t)e * 4096 + rgrp * 64 + cb;

    // warm the tail of this edge's 8KB matrix (lowers to global_prefetch_b8)
    __builtin_prefetch(wp + 2048, 0, 1);
    __builtin_prefetch(wp + 3072, 0, 1);

    float acc[8] = {0.f, 0.f, 0.f, 0.f, 0.f, 0.f, 0.f, 0.f};
#pragma unroll
    for (int it = 0; it < 16; ++it) {
        uint4 w = *(const uint4*)(wp + it * 256);   // rows it*4+rgrp, 16B per lane
        float xi = xs[wave][it * 4 + rgrp];
        acc[0] += xi * bf16lo(w.x);  acc[1] += xi * bf16hi(w.x);
        acc[2] += xi * bf16lo(w.y);  acc[3] += xi * bf16hi(w.y);
        acc[4] += xi * bf16lo(w.z);  acc[5] += xi * bf16hi(w.z);
        acc[6] += xi * bf16lo(w.w);  acc[7] += xi * bf16hi(w.w);
    }
#pragma unroll
    for (int j = 0; j < 8; ++j) {
        acc[j] += __shfl_xor(acc[j], 8);
        acc[j] += __shfl_xor(acc[j], 16);
    }
    if (rgrp == 0) {                      // lanes 0..7 hold the full column sums
        float* dst = agg + (size_t)tgt * 64 + cb;
#pragma unroll
        for (int j = 0; j < 8; ++j) atomicAdd(&dst[j], acc[j]);
    }
}

// ---------- m = relu(agg/deg + conv_b), in place ----------
__global__ void k_msg(float* __restrict__ agg, const float* __restrict__ deg,
                      const float* __restrict__ cb) {
    int t = blockIdx.x * blockDim.x + threadIdx.x;
    if (t >= N_NODES * DIM) return;
    int n = t >> 6, d = t & 63;
    float v = agg[t] / fmaxf(deg[n], 1.0f) + cb[d];
    agg[t] = fmaxf(v, 0.0f);
}

// ---------- GRU GEMMs: G1 = m@Wih^T+bih ; G2 = h@Whh^T+bhh (weights pre-transposed) ----------
__global__ void k_gru_gemm(const float* __restrict__ m, const float* __restrict__ h,
                           const float* __restrict__ wTih, const float* __restrict__ wThh,
                           const float* __restrict__ bih, const float* __restrict__ bhh,
                           float* __restrict__ G1, float* __restrict__ G2) {
    const int n = blockIdx.x, j = threadIdx.x;    // blockDim = 192
    float g1 = bih[j], g2 = bhh[j];
#pragma unroll 8
    for (int k = 0; k < 64; ++k) {
        float mk = m[(size_t)n * 64 + k];         // scalar broadcast
        float hk = h[(size_t)n * 64 + k];
        g1 += mk * wTih[k * 192 + j];             // coalesced
        g2 += hk * wThh[k * 192 + j];
    }
    G1[(size_t)n * 192 + j] = g1;
    G2[(size_t)n * 192 + j] = g2;
}

__global__ void k_gru_gate(const float* __restrict__ G1, const float* __restrict__ G2,
                           float* __restrict__ h) {
    int t = blockIdx.x * blockDim.x + threadIdx.x;
    if (t >= N_NODES * DIM) return;
    int n = t >> 6, d = t & 63;
    size_t b = (size_t)n * 192;
    float rg = sigm(G1[b + d] + G2[b + d]);
    float zg = sigm(G1[b + 64 + d] + G2[b + 64 + d]);
    float ng = tanhf(G1[b + 128 + d] + rg * G2[b + 128 + d]);
    float hp = h[t];
    h[t] = (1.0f - zg) * ng + zg * hp;
}

// ---------- Set2Set LSTM step (one block per batch element) ----------
__global__ void k_lstm(const float* __restrict__ qstar, float* __restrict__ hl, float* __restrict__ cl,
                       const float* __restrict__ wTih, const float* __restrict__ wThh,
                       const float* __restrict__ bih, const float* __restrict__ bhh) {
    __shared__ float gs[256];
    const int b = blockIdx.x, j = threadIdx.x;    // blockDim = 256
    float g = bih[j] + bhh[j];
    for (int k = 0; k < 128; ++k) g += qstar[(size_t)b * 128 + k] * wTih[k * 256 + j];
    for (int k = 0; k < 64; ++k)  g += hl[(size_t)b * 64 + k]   * wThh[k * 256 + j];
    gs[j] = g;
    __syncthreads();
    if (j < 64) {
        float ig = sigm(gs[j]), fg = sigm(gs[64 + j]);
        float gg = tanhf(gs[128 + j]), og = sigm(gs[192 + j]);
        float c = fg * cl[(size_t)b * 64 + j] + ig * gg;
        cl[(size_t)b * 64 + j] = c;
        hl[(size_t)b * 64 + j] = og * tanhf(c);
    }
}

// ---------- attention: logits + segment max ----------
__global__ void k_att_e(const float* __restrict__ out, const float* __restrict__ hl,
                        const int* __restrict__ batch, float* __restrict__ ev,
                        unsigned* __restrict__ emaxU) {
    int t = blockIdx.x * blockDim.x + threadIdx.x;
    if (t >= N_NODES) return;
    int b = batch[t];
    float acc = 0.0f;
#pragma unroll 8
    for (int k = 0; k < 64; ++k) acc += out[(size_t)t * 64 + k] * hl[(size_t)b * 64 + k];
    ev[t] = acc;
    atomicMax(&emaxU[b], f32_ord(acc));
}

__global__ void k_att_ex(const float* __restrict__ ev, const unsigned* __restrict__ emaxU,
                         const int* __restrict__ batch, float* __restrict__ exv,
                         float* __restrict__ denom) {
    int t = blockIdx.x * blockDim.x + threadIdx.x;
    if (t >= N_NODES) return;
    int b = batch[t];
    float ex = __expf(ev[t] - ord_f32(emaxU[b]));
    exv[t] = ex;
    atomicAdd(&denom[b], ex);
}

__global__ void k_att_rvec(const float* __restrict__ exv, const float* __restrict__ denom,
                           const float* __restrict__ out, const int* __restrict__ batch,
                           float* __restrict__ rvec) {
    int t = blockIdx.x * blockDim.x + threadIdx.x;
    if (t >= N_NODES * DIM) return;
    int n = t >> 6, d = t & 63;
    int b = batch[n];
    float a = exv[n] / (denom[b] + 1e-16f);
    atomicAdd(&rvec[(size_t)b * 64 + d], a * out[t]);
}

__global__ void k_qstar(const float* __restrict__ hl, const float* __restrict__ rvec,
                        float* __restrict__ qstar) {
    int t = blockIdx.x * blockDim.x + threadIdx.x;   // B*128
    int b = t >> 7, j = t & 127;
    qstar[t] = (j < 64) ? hl[(size_t)b * 64 + j] : rvec[(size_t)b * 64 + (j - 64)];
}

// ---------- final head: y = relu(q*@lin1^T+b1)@lin2^T+b2 ----------
__global__ void k_final(const float* __restrict__ qstar, const float* __restrict__ w1,
                        const float* __restrict__ b1, const float* __restrict__ w2,
                        const float* __restrict__ b2, float* __restrict__ y) {
    __shared__ float q1[64];
    const int b = blockIdx.x, d = threadIdx.x;   // blockDim = 64
    float acc = b1[d];
    for (int k = 0; k < 128; ++k) acc += qstar[(size_t)b * 128 + k] * w1[d * 128 + k];
    q1[d] = fmaxf(acc, 0.0f);
    __syncthreads();
    if (d < NOUT) {
        float a2 = b2[d];
#pragma unroll
        for (int k = 0; k < 64; ++k) a2 += q1[k] * w2[d * 64 + k];
        y[(size_t)b * NOUT + d] = a2;
    }
}

// =====================================================================
extern "C" void kernel_launch(void* const* d_in, const int* in_sizes, int n_in,
                              void* d_out, int out_size, void* d_ws, size_t ws_size,
                              hipStream_t stream) {
    const float* x        = (const float*)d_in[0];
    const float* ea       = (const float*)d_in[1];
    const int*   ei       = (const int*)  d_in[2];
    const int*   batch    = (const int*)  d_in[3];
    const float* lin0_w   = (const float*)d_in[4];
    const float* lin0_b   = (const float*)d_in[5];
    const float* en_w1    = (const float*)d_in[6];
    const float* en_b1    = (const float*)d_in[7];
    const float* en_w2    = (const float*)d_in[8];
    const float* en_b2    = (const float*)d_in[9];
    const float* conv_b   = (const float*)d_in[10];
    const float* gru_w_ih = (const float*)d_in[11];
    const float* gru_w_hh = (const float*)d_in[12];
    const float* gru_b_ih = (const float*)d_in[13];
    const float* gru_b_hh = (const float*)d_in[14];
    const float* lstm_w_ih= (const float*)d_in[15];
    const float* lstm_w_hh= (const float*)d_in[16];
    const float* lstm_b_ih= (const float*)d_in[17];
    const float* lstm_b_hh= (const float*)d_in[18];
    const float* lin1_w   = (const float*)d_in[19];
    const float* lin1_b   = (const float*)d_in[20];
    const float* lin2_w   = (const float*)d_in[21];
    const float* lin2_b   = (const float*)d_in[22];
    float* y = (float*)d_out;

    // ---- workspace carve ----
    size_t off = 0;
    auto carve = [&](size_t bytes) -> void* {
        off = (off + 255) & ~(size_t)255;
        void* p = (char*)d_ws + off;
        off += bytes;
        return p;
    };
    unsigned short* ew      = (unsigned short*)carve((size_t)N_EDGES * 4096 * 2);   // 491.5 MB bf16
    unsigned short* hidpack = (unsigned short*)carve((size_t)N_EDGES * 128 * 2);
    unsigned short* w2pack  = (unsigned short*)carve((size_t)4096 * 128 * 2);
    float* hbuf  = (float*)carve((size_t)N_NODES * 64 * 4);     // out == h
    float* agg   = (float*)carve((size_t)N_NODES * 64 * 4);     // agg -> m in place
    float* G1    = (float*)carve((size_t)N_NODES * 192 * 4);
    float* G2    = (float*)carve((size_t)N_NODES * 192 * 4);
    float* deg   = (float*)carve((size_t)N_NODES * 4);
    float* wTih  = (float*)carve((size_t)64 * 192 * 4);
    float* wThh  = (float*)carve((size_t)64 * 192 * 4);
    float* wTihL = (float*)carve((size_t)128 * 256 * 4);
    float* wThhL = (float*)carve((size_t)64 * 256 * 4);
    float* qstar = (float*)carve((size_t)NBATCH * 128 * 4);
    float* hl    = (float*)carve((size_t)NBATCH * 64 * 4);
    float* cl    = (float*)carve((size_t)NBATCH * 64 * 4);
    float* ev    = (float*)carve((size_t)N_NODES * 4);
    float* exv   = (float*)carve((size_t)N_NODES * 4);
    unsigned* emaxU = (unsigned*)carve((size_t)NBATCH * 4);
    float* denom = (float*)carve((size_t)NBATCH * 4);
    float* rvec  = (float*)carve((size_t)NBATCH * 64 * 4);
    (void)ws_size; (void)in_sizes; (void)n_in; (void)out_size;

    const int T = 256;
    auto cdiv = [](int a, int b) { return (a + b - 1) / b; };

    // ---- one-time packs / transposes ----
    k_pack_w2<<<(4096 * 128) / T, T, 0, stream>>>(en_w2, w2pack);
    k_transpose<<<cdiv(192 * 64, T), T, 0, stream>>>(gru_w_ih, wTih, 192, 64);
    k_transpose<<<cdiv(192 * 64, T), T, 0, stream>>>(gru_w_hh, wThh, 192, 64);
    k_transpose<<<cdiv(256 * 128, T), T, 0, stream>>>(lstm_w_ih, wTihL, 256, 128);
    k_transpose<<<cdiv(256 * 64, T), T, 0, stream>>>(lstm_w_hh, wThhL, 256, 64);

    // ---- edge network (WMMA) ----
    k_hid_pack<<<(N_EDGES * 128) / T, T, 0, stream>>>(ea, en_w1, en_b1, hidpack);
    dim3 gg(N_EDGES / 16, 256 / 8);
    k_gemm_ew<<<gg, 256, 0, stream>>>(hidpack, w2pack, en_b2, ew);

    // ---- lin0 + degree ----
    k_lin0<<<cdiv(N_NODES * 64, T), T, 0, stream>>>(x, lin0_w, lin0_b, hbuf);
    k_fill_f32<<<cdiv(N_NODES, T), T, 0, stream>>>(deg, 0.0f, N_NODES);
    k_deg<<<cdiv(N_EDGES, T), T, 0, stream>>>(ei, deg);

    // ---- 6 message-passing steps ----
    for (int s = 0; s < STEPS; ++s) {
        k_fill_f32<<<cdiv(N_NODES * 64, T), T, 0, stream>>>(agg, 0.0f, N_NODES * 64);
        k_edge_matvec<<<N_EDGES / 8, 256, 0, stream>>>(ew, hbuf, ei, agg);
        k_msg<<<cdiv(N_NODES * 64, T), T, 0, stream>>>(agg, deg, conv_b);
        k_gru_gemm<<<N_NODES, 192, 0, stream>>>(agg, hbuf, wTih, wThh, gru_b_ih, gru_b_hh, G1, G2);
        k_gru_gate<<<cdiv(N_NODES * 64, T), T, 0, stream>>>(G1, G2, hbuf);
    }

    // ---- Set2Set ----
    k_fill_f32<<<cdiv(NBATCH * 128, T), T, 0, stream>>>(qstar, 0.0f, NBATCH * 128);
    k_fill_f32<<<cdiv(NBATCH * 64, T), T, 0, stream>>>(hl, 0.0f, NBATCH * 64);
    k_fill_f32<<<cdiv(NBATCH * 64, T), T, 0, stream>>>(cl, 0.0f, NBATCH * 64);
    for (int s = 0; s < STEPS; ++s) {
        k_lstm<<<NBATCH, 256, 0, stream>>>(qstar, hl, cl, wTihL, wThhL, lstm_b_ih, lstm_b_hh);
        k_fill_u32<<<cdiv(NBATCH, T), T, 0, stream>>>(emaxU, 0u, NBATCH);
        k_fill_f32<<<cdiv(NBATCH, T), T, 0, stream>>>(denom, 0.0f, NBATCH);
        k_fill_f32<<<cdiv(NBATCH * 64, T), T, 0, stream>>>(rvec, 0.0f, NBATCH * 64);
        k_att_e<<<cdiv(N_NODES, T), T, 0, stream>>>(hbuf, hl, batch, ev, emaxU);
        k_att_ex<<<cdiv(N_NODES, T), T, 0, stream>>>(ev, emaxU, batch, exv, denom);
        k_att_rvec<<<cdiv(N_NODES * 64, T), T, 0, stream>>>(exv, denom, hbuf, batch, rvec);
        k_qstar<<<(NBATCH * 128) / T, T, 0, stream>>>(hl, rvec, qstar);
    }

    // ---- output head ----
    k_final<<<NBATCH, 64, 0, stream>>>(qstar, lin1_w, lin1_b, lin2_w, lin2_b, y);
}
